// AttentionIn_63711544869145
// MI455X (gfx1250) — matrix operand order repaired and checked by reading, not compiled, not run on hardware
//
#include <hip/hip_runtime.h>
#include <hip/hip_bf16.h>

// ---- sizes (compile-time, from reference) ----
#define N_HEADS    16
#define D_MODEL    2048
#define D_HEAD     128
#define ROT_DIM    64
#define BATCH      4
#define SEQ        2048
#define M_TOTAL    (BATCH * SEQ)          // 8192 rows
#define N_TOTAL    (N_HEADS * D_HEAD)     // 2048 cols per projection
#define K_TOTAL    D_MODEL                // 2048
#define OUT_ELEMS  ((size_t)M_TOTAL * N_TOTAL)   // 16,777,216 per tensor

// ---- workspace layout (bytes) ----
#define WS_A_HI    (0ULL)
#define WS_A_LO    (33554432ULL)                  // 32 MB
#define WS_B_BASE  (67108864ULL)                  // 64 MB
#define WS_B_STEP  (16777216ULL)                  // 16 MB per projection (hi+lo)
#define B_ELEMS    ((size_t)N_TOTAL * K_TOTAL)    // 4,194,304 per matrix
#define WS_NEEDED  (117440512ULL)                 // 112 MB

typedef __attribute__((ext_vector_type(16))) __bf16          v16bf;
typedef __attribute__((ext_vector_type(8)))  float           v8f;
typedef __attribute__((ext_vector_type(16))) unsigned short  v16u;
typedef __attribute__((ext_vector_type(8)))  unsigned short  v8u;
typedef __attribute__((ext_vector_type(4)))  unsigned short  v4u;

// ---- bf16 split helpers ----
// Truncation split: hi = top16(f), lo = top16(f - hi).  |err| < 2^-16 * |f|
__device__ __forceinline__ unsigned short f2bf_trunc(float f) {
    return (unsigned short)(__float_as_uint(f) >> 16);
}
__device__ __forceinline__ float bf2f(unsigned short h) {
    return __uint_as_float(((unsigned int)h) << 16);
}
// RNE split (used by fallback kernel)
__device__ __forceinline__ unsigned short f2bf_rne(float f) {
    unsigned int u = __float_as_uint(f);
    unsigned int r = u + 0x7FFFu + ((u >> 16) & 1u);
    return (unsigned short)(r >> 16);
}

__device__ __forceinline__ v8f wmma_bf16(v16bf a, v16bf b, v8f c) {
    return __builtin_amdgcn_wmma_f32_16x16x32_bf16(
        false, a, false, b, (short)0, c, false, false);
}

// ---- CDNA5 async global->LDS (ASYNCcnt-tracked) ----
__device__ __forceinline__ void async_g2l_b128(unsigned lds_addr, unsigned voff,
                                               unsigned long long sbase) {
    asm volatile("global_load_async_to_lds_b128 %0, %1, %2"
                 :: "v"(lds_addr), "v"(voff), "s"(sbase)
                 : "memory");
}

// ---------------------------------------------------------------------------
// Residual pass-through
// ---------------------------------------------------------------------------
__global__ __launch_bounds__(256) void residual_copy_kernel(
    const float4* __restrict__ in, float4* __restrict__ out, int n4)
{
    int stride = gridDim.x * blockDim.x;
    for (int i = blockIdx.x * blockDim.x + threadIdx.x; i < n4; i += stride)
        out[i] = in[i];
}

// ---------------------------------------------------------------------------
// Preprocess: split x (fp32 [8192][2048]) -> Ahi/Alo bf16, same layout.
// ---------------------------------------------------------------------------
__global__ __launch_bounds__(256) void prep_x_kernel(
    const float* __restrict__ x, unsigned short* __restrict__ ahi,
    unsigned short* __restrict__ alo, int n4)
{
    int stride = gridDim.x * blockDim.x;
    for (int i = blockIdx.x * blockDim.x + threadIdx.x; i < n4; i += stride) {
        const float4 f = reinterpret_cast<const float4*>(x)[i];
        v4u h, l;
        float fv[4] = {f.x, f.y, f.z, f.w};
        #pragma unroll
        for (int j = 0; j < 4; ++j) {
            unsigned short hh = f2bf_trunc(fv[j]);
            h[j] = hh;
            l[j] = f2bf_trunc(fv[j] - bf2f(hh));
        }
        reinterpret_cast<v4u*>(ahi)[i] = h;
        reinterpret_cast<v4u*>(alo)[i] = l;
    }
}

// ---------------------------------------------------------------------------
// Preprocess: W [16][2048][128] fp32 -> Bhi/Blo bf16 in [n][k] layout
// (k contiguous).  out[n*2048 + k] = W[n>>7][k][n&127]
// ---------------------------------------------------------------------------
__global__ __launch_bounds__(256) void prep_w_kernel(
    const float* __restrict__ Wq, const float* __restrict__ Wk,
    const float* __restrict__ Wv, char* __restrict__ ws)
{
    const int z = blockIdx.z;
    const float* __restrict__ W = (z == 0) ? Wq : ((z == 1) ? Wk : Wv);
    unsigned short* __restrict__ bh =
        (unsigned short*)(ws + WS_B_BASE + (size_t)z * WS_B_STEP);
    unsigned short* __restrict__ bl = bh + B_ELEMS;

    const int n4 = (int)(B_ELEMS / 4);
    int stride = gridDim.x * blockDim.x;
    for (int i = blockIdx.x * blockDim.x + threadIdx.x; i < n4; i += stride) {
        const int idx = i * 4;                 // 4 consecutive k, fixed n
        const int k = idx & (K_TOTAL - 1);
        const int n = idx >> 11;
        const float* src = W + (size_t)(n >> 7) * (K_TOTAL * D_HEAD)
                             + (n & 127) + (size_t)k * D_HEAD;
        v4u h, l;
        #pragma unroll
        for (int j = 0; j < 4; ++j) {
            float f = src[(size_t)j * D_HEAD];
            unsigned short hh = f2bf_trunc(f);
            h[j] = hh;
            l[j] = f2bf_trunc(f - bf2f(hh));
        }
        reinterpret_cast<v4u*>(bh)[i] = h;
        reinterpret_cast<v4u*>(bl)[i] = l;
    }
}

// ---------------------------------------------------------------------------
// Fast GEMM: pre-split bf16 operands, async double-buffered B tiles in LDS.
//   grid = (64, 32, 3), block = 256 (8 waves), wave tile 16x64, K step 32.
// ---------------------------------------------------------------------------
__global__ __launch_bounds__(256) void qkv_rope_wmma_fast_kernel(
    const char* __restrict__ ws,
    const float* __restrict__ bq, const float* __restrict__ bk,
    const float* __restrict__ bv, float* __restrict__ out)
{
    const int z = blockIdx.z;
    const float* __restrict__ bias = (z == 0) ? bq : ((z == 1) ? bk : bv);
    float* __restrict__ o = out + (size_t)(1 + z) * OUT_ELEMS;

    const unsigned short* __restrict__ Ahi = (const unsigned short*)(ws + WS_A_HI);
    const unsigned short* __restrict__ Alo = (const unsigned short*)(ws + WS_A_LO);
    const unsigned short* __restrict__ Bhi =
        (const unsigned short*)(ws + WS_B_BASE + (size_t)z * WS_B_STEP);
    const unsigned short* __restrict__ Blo = Bhi + B_ELEMS;

    // double-buffered B tile: [buf][hi/lo][n(64)][k(32)] bf16 = 16 KB
    __shared__ __align__(16) unsigned short lb[2][2][64 * 32];

    const int tid   = threadIdx.x;
    const int lane  = tid & 31;
    const int wave  = tid >> 5;
    const int lrow  = lane & 15;
    const int lhalf = lane >> 4;

    const int m0 = blockIdx.x * 128 + wave * 16;
    const int n0 = blockIdx.y * 64;

    // A addressing: lane = row m0+lrow, chunk [lhalf*16,+16)
    const unsigned short* __restrict__ ah_ptr =
        Ahi + (size_t)(m0 + lrow) * K_TOTAL + lhalf * 16;
    const unsigned short* __restrict__ al_ptr =
        Alo + (size_t)(m0 + lrow) * K_TOTAL + lhalf * 16;

    // async copy addressing: thread -> (n_local = tid>>2, 16B chunk q = tid&3)
    const int nl = tid >> 2;
    const int q  = tid & 3;
    const unsigned voff = (unsigned)(nl * (K_TOTAL * 2) + q * 16); // bytes
    const unsigned lds_base = (unsigned)(size_t)&lb[0][0][0];
    const unsigned lds_tile_off = (unsigned)((nl * 32 + q * 8) * 2);
    const unsigned long long gb_hi =
        (unsigned long long)(size_t)(Bhi + (size_t)n0 * K_TOTAL);
    const unsigned long long gb_lo =
        (unsigned long long)(size_t)(Blo + (size_t)n0 * K_TOTAL);

    v8f acc[4];
    #pragma unroll
    for (int j = 0; j < 4; ++j)
        #pragma unroll
        for (int e = 0; e < 8; ++e) acc[j][e] = 0.0f;

    // prologue: issue buffer 0 (k0 = 0)
    async_g2l_b128(lds_base + 0 * 8192 + 0 * 4096 + lds_tile_off, voff, gb_hi);
    async_g2l_b128(lds_base + 0 * 8192 + 1 * 4096 + lds_tile_off, voff, gb_lo);

    const int NI = K_TOTAL / 32;   // 64
    for (int i = 0; i < NI; ++i) {
        const int k0  = i * 32;
        const int cur = i & 1;

        if (i + 1 < NI) {
            const unsigned nb = (unsigned)((i + 1) & 1) * 8192u;
            const unsigned kb = (unsigned)(k0 + 32) * 2u;   // byte advance in k
            async_g2l_b128(lds_base + nb + 0 * 4096 + lds_tile_off, voff, gb_hi + kb);
            async_g2l_b128(lds_base + nb + 1 * 4096 + lds_tile_off, voff, gb_lo + kb);
            asm volatile("s_wait_asynccnt 0x2" ::: "memory");
        } else {
            asm volatile("s_wait_asynccnt 0x0" ::: "memory");
        }
        __syncthreads();

        // ---- A fragments (pre-split bf16, contiguous 16B loads) ----
        v8u ah0 = *reinterpret_cast<const v8u*>(ah_ptr + k0);
        v8u ah1 = *reinterpret_cast<const v8u*>(ah_ptr + k0 + 8);
        v8u al0 = *reinterpret_cast<const v8u*>(al_ptr + k0);
        v8u al1 = *reinterpret_cast<const v8u*>(al_ptr + k0 + 8);
        v16u ahu = __builtin_shufflevector(ah0, ah1, 0,1,2,3,4,5,6,7,8,9,10,11,12,13,14,15);
        v16u alu = __builtin_shufflevector(al0, al1, 0,1,2,3,4,5,6,7,8,9,10,11,12,13,14,15);
        v16bf Ahi_f = __builtin_bit_cast(v16bf, ahu);
        v16bf Alo_f = __builtin_bit_cast(v16bf, alu);

        // ---- 4 N-tiles x 3 WMMAs ----
        const unsigned short* lbh = &lb[cur][0][0];
        const unsigned short* lbl = &lb[cur][1][0];
        #pragma unroll
        for (int j = 0; j < 4; ++j) {
            const int base = (j * 16 + lrow) * 32 + lhalf * 16;
            v16u bhu = *reinterpret_cast<const v16u*>(&lbh[base]);
            v16u blu = *reinterpret_cast<const v16u*>(&lbl[base]);
            v16bf Bh = __builtin_bit_cast(v16bf, bhu);
            v16bf Bl = __builtin_bit_cast(v16bf, blu);
            acc[j] = wmma_bf16(Alo_f, Bh, acc[j]);
            acc[j] = wmma_bf16(Ahi_f, Bl, acc[j]);
            acc[j] = wmma_bf16(Ahi_f, Bh, acc[j]);
        }
        __syncthreads();
    }

    // ---- epilogue: bias, rotary (Q/K, e<64), store ----
    const bool isQK = (z < 2);
    #pragma unroll
    for (int j = 0; j < 4; ++j) {
        const int n = n0 + j * 16 + lrow;
        const float bb = bias[n];
        const int e = n & 127;
        const bool rot = isQK && (e < ROT_DIM);     // wave-uniform
        float inv_freq = 0.0f;
        if (rot)
            inv_freq = __powf(10000.0f, -(float)(e >> 1) * (1.0f / 32.0f));

        #pragma unroll
        for (int v = 0; v < 8; ++v) {
            const int m = m0 + v + lhalf * 8;
            float val = acc[j][v] + bb;
            if (rot) {
                const int pos = m & (SEQ - 1);
                float s, c;
                __sincosf((float)pos * inv_freq, &s, &c);
                const float other = __shfl_xor(val, 1, 32);
                val = ((n & 1) == 0) ? (val * c - other * s)
                                     : (val * c + other * s);
            }
            o[(size_t)m * N_TOTAL + n] = val;
        }
    }
}

// ---------------------------------------------------------------------------
// Fallback fused kernel (no workspace needed).
// ---------------------------------------------------------------------------
__global__ __launch_bounds__(256) void qkv_rope_wmma_kernel(
    const float* __restrict__ x,
    const float* __restrict__ Wq, const float* __restrict__ Wk, const float* __restrict__ Wv,
    const float* __restrict__ bq, const float* __restrict__ bk, const float* __restrict__ bv,
    float* __restrict__ out)
{
    const int z = blockIdx.z;
    const float* __restrict__ W    = (z == 0) ? Wq : ((z == 1) ? Wk : Wv);
    const float* __restrict__ bias = (z == 0) ? bq : ((z == 1) ? bk : bv);
    float* __restrict__ o = out + (size_t)(1 + z) * OUT_ELEMS;

    __shared__ __align__(16) unsigned short lbh[64 * 32];
    __shared__ __align__(16) unsigned short lbl[64 * 32];

    const int tid   = threadIdx.x;
    const int lane  = tid & 31;
    const int wave  = tid >> 5;
    const int lrow  = lane & 15;
    const int lhalf = lane >> 4;

    const int m0 = blockIdx.x * 128 + wave * 16;
    const int n0 = blockIdx.y * 64;

    const float* __restrict__ aptr = x + (size_t)(m0 + lrow) * K_TOTAL + lhalf * 16;

    const int nl = tid & 63;
    const int kg = tid >> 6;
    const int ncol = n0 + nl;
    const float* __restrict__ wptr =
        W + (size_t)(ncol >> 7) * (K_TOTAL * D_HEAD) + (ncol & 127)
          + (size_t)(kg * 8) * D_HEAD;

    v8f acc[4];
    #pragma unroll
    for (int j = 0; j < 4; ++j)
        #pragma unroll
        for (int e = 0; e < 8; ++e) acc[j][e] = 0.0f;

    for (int k0 = 0; k0 < K_TOTAL; k0 += 32) {
        v8u hvec, lvec;
        #pragma unroll
        for (int kk = 0; kk < 8; ++kk) {
            float w = wptr[(size_t)(k0 + kk) * D_HEAD];
            unsigned short h = f2bf_rne(w);
            hvec[kk] = h;
            lvec[kk] = f2bf_rne(w - bf2f(h));
        }
        *reinterpret_cast<v8u*>(&lbh[nl * 32 + kg * 8]) = hvec;
        *reinterpret_cast<v8u*>(&lbl[nl * 32 + kg * 8]) = lvec;
        __syncthreads();

        float av[16];
        {
            const float4 t0 = *reinterpret_cast<const float4*>(aptr + k0);
            const float4 t1 = *reinterpret_cast<const float4*>(aptr + k0 + 4);
            const float4 t2 = *reinterpret_cast<const float4*>(aptr + k0 + 8);
            const float4 t3 = *reinterpret_cast<const float4*>(aptr + k0 + 12);
            av[0]=t0.x; av[1]=t0.y; av[2]=t0.z;  av[3]=t0.w;
            av[4]=t1.x; av[5]=t1.y; av[6]=t1.z;  av[7]=t1.w;
            av[8]=t2.x; av[9]=t2.y; av[10]=t2.z; av[11]=t2.w;
            av[12]=t3.x; av[13]=t3.y; av[14]=t3.z; av[15]=t3.w;
        }
        v16u ahu, alu;
        #pragma unroll
        for (int i = 0; i < 16; ++i) {
            unsigned short h = f2bf_rne(av[i]);
            ahu[i] = h;
            alu[i] = f2bf_rne(av[i] - bf2f(h));
        }
        v16bf Ahi = __builtin_bit_cast(v16bf, ahu);
        v16bf Alo = __builtin_bit_cast(v16bf, alu);

        #pragma unroll
        for (int j = 0; j < 4; ++j) {
            const int base = (j * 16 + lrow) * 32 + lhalf * 16;
            v16u bhu = *reinterpret_cast<const v16u*>(&lbh[base]);
            v16u blu = *reinterpret_cast<const v16u*>(&lbl[base]);
            v16bf Bh = __builtin_bit_cast(v16bf, bhu);
            v16bf Bl = __builtin_bit_cast(v16bf, blu);
            acc[j] = wmma_bf16(Alo, Bh, acc[j]);
            acc[j] = wmma_bf16(Ahi, Bl, acc[j]);
            acc[j] = wmma_bf16(Ahi, Bh, acc[j]);
        }
        __syncthreads();
    }

    const bool isQK = (z < 2);
    #pragma unroll
    for (int j = 0; j < 4; ++j) {
        const int n = n0 + j * 16 + lrow;
        const float bb = bias[n];
        const int e = n & 127;
        const bool rot = isQK && (e < ROT_DIM);
        float inv_freq = 0.0f;
        if (rot)
            inv_freq = __powf(10000.0f, -(float)(e >> 1) * (1.0f / 32.0f));

        #pragma unroll
        for (int v = 0; v < 8; ++v) {
            const int m = m0 + v + lhalf * 8;
            float val = acc[j][v] + bb;
            if (rot) {
                const int pos = m & (SEQ - 1);
                float s, c;
                __sincosf((float)pos * inv_freq, &s, &c);
                const float other = __shfl_xor(val, 1, 32);
                val = ((n & 1) == 0) ? (val * c - other * s)
                                     : (val * c + other * s);
            }
            o[(size_t)m * N_TOTAL + n] = val;
        }
    }
}

extern "C" void kernel_launch(void* const* d_in, const int* in_sizes, int n_in,
                              void* d_out, int out_size, void* d_ws, size_t ws_size,
                              hipStream_t stream) {
    const float* residual = (const float*)d_in[0];
    const float* x        = (const float*)d_in[1];
    const float* Wq       = (const float*)d_in[2];
    const float* Wk       = (const float*)d_in[3];
    const float* Wv       = (const float*)d_in[4];
    const float* bq       = (const float*)d_in[5];
    const float* bk       = (const float*)d_in[6];
    const float* bv       = (const float*)d_in[7];
    float* out = (float*)d_out;

    residual_copy_kernel<<<4096, 256, 0, stream>>>(
        (const float4*)residual, (float4*)out, (int)(OUT_ELEMS / 4));

    dim3 grid(M_TOTAL / 128, N_TOTAL / 64, 3);

    if (ws_size >= WS_NEEDED) {
        char* ws = (char*)d_ws;
        unsigned short* ahi = (unsigned short*)(ws + WS_A_HI);
        unsigned short* alo = (unsigned short*)(ws + WS_A_LO);
        prep_x_kernel<<<4096, 256, 0, stream>>>(x, ahi, alo, (int)(OUT_ELEMS / 4));
        prep_w_kernel<<<dim3(2048, 1, 3), 256, 0, stream>>>(Wq, Wk, Wv, ws);
        qkv_rope_wmma_fast_kernel<<<grid, 256, 0, stream>>>(ws, bq, bk, bv, out);
    } else {
        qkv_rope_wmma_kernel<<<grid, 256, 0, stream>>>(
            x, Wq, Wk, Wv, bq, bk, bv, out);
    }
}
